// AttentionHead_41386304864763
// MI455X (gfx1250) — matrix operand order repaired and checked by reading, not compiled, not run on hardware
//
#include <hip/hip_runtime.h>
#include <hip/hip_bf16.h>

typedef __bf16 bf16_t;
typedef __attribute__((ext_vector_type(16))) __bf16 v16bf;
typedef __attribute__((ext_vector_type(8)))  __bf16 v8bf;
typedef __attribute__((ext_vector_type(4)))  __bf16 v4bf;
typedef __attribute__((ext_vector_type(8)))  float  v8f;

#define SLEN 4096
#define HID  4096
#define HD   128
#define BATCH 4

// native fp32 -> bf16 (compiler emits v_cvt_pk_bf16_f32, RTNE)
static __device__ __forceinline__ bf16_t f2bf(float f) {
    return static_cast<bf16_t>(f);
}

// ---------------------------------------------------------------------------
// Kernel 0: one-time fp32 -> bf16 conversion of the three weight matrices.
// ---------------------------------------------------------------------------
__global__ __launch_bounds__(256) void wcvt_k(
    const float* __restrict__ wq, const float* __restrict__ wk,
    const float* __restrict__ wv, bf16_t* __restrict__ wb)
{
    const float* src = (blockIdx.y == 0) ? wq : (blockIdx.y == 1) ? wk : wv;
    bf16_t* dst = wb + (size_t)blockIdx.y * HD * HID;
    int i = blockIdx.x * blockDim.x + threadIdx.x;       // float4 index
    float4 v = ((const float4*)src)[i];
    v4bf p;
    p[0] = f2bf(v.x); p[1] = f2bf(v.y); p[2] = f2bf(v.z); p[3] = f2bf(v.w);
    ((v4bf*)dst)[i] = p;
}

// ---------------------------------------------------------------------------
// Kernel 1: fused QKV projection + bias + RoPE.
// grid = 3*1024 blocks, 1 wave each; proj = blockIdx.x % 3 so the three
// projections of one x row-tile are dispatched adjacently and share the
// 256 KB x tile through L2 (x HBM traffic ~256 MB instead of 768 MB).
// Software-pipelined: B-fragments double-buffered in registers (ping-pong),
// A-tile fp32 loads issued one stage ahead.
// ---------------------------------------------------------------------------
__global__ __launch_bounds__(32) void qkv_rope_k(
    const float* __restrict__ x,
    const bf16_t* __restrict__ wb,    // [3][HD][HID] bf16
    const float* __restrict__ bq, const float* __restrict__ bk,
    const float* __restrict__ bv,
    const float* __restrict__ rope_cos, const float* __restrict__ rope_sin,
    float* __restrict__ out,          // d_out: [attn | k | v], each B*S*HD f32
    bf16_t* __restrict__ qb,          // ws: q (scaled) bf16 row-major [m][d]
    bf16_t* __restrict__ kb,          // ws: k bf16 row-major [m][d]
    bf16_t* __restrict__ vtb)         // ws: v bf16 transposed [b][d][s]
{
    const int lane = threadIdx.x;
    const int lnm  = lane & 15;
    const int half = lane >> 4;
    const int tile = blockIdx.x / 3;
    const int proj = blockIdx.x - 3 * tile;   // 0=q 1=k 2=v (fastest-varying)
    const int m0   = tile * 16;               // global row tile
    const int bidx = m0 / SLEN;
    const int s_base = m0 - bidx * SLEN;

    const bf16_t* w = wb + (size_t)proj * HD * HID;
    const float* bias = (proj == 0) ? bq : (proj == 1) ? bk : bv;

    v8f zero = {0.f,0.f,0.f,0.f,0.f,0.f,0.f,0.f};
    v8f acc[8];
    #pragma unroll
    for (int t = 0; t < 8; ++t) acc[t] = zero;

    const float*  xrow = x + (size_t)(m0 + lnm) * HID + half * 16;
    const bf16_t* wrow = w + (size_t)lnm * HID + half * 16;

    // ---- software pipeline prologue: load stage-0 operands ----
    float4 a0, a1, a2, a3;
    {
        const float4* xp = (const float4*)(xrow);
        a0 = xp[0]; a1 = xp[1]; a2 = xp[2]; a3 = xp[3];
    }
    v16bf bbuf0[8], bbuf1[8];
    #pragma unroll
    for (int t = 0; t < 8; ++t)
        bbuf0[t] = *(const v16bf*)(wrow + (size_t)(t*16) * HID);

    // one pipeline stage: convert current A, issue loads for `kload`,
    // run 8 WMMAs on BC, leaving new operands in a0..a3 / BN.
    auto stage = [&](int kload, v16bf* BC, v16bf* BN) {
        v16bf af;
        af[0]=f2bf(a0.x); af[1]=f2bf(a0.y); af[2]=f2bf(a0.z); af[3]=f2bf(a0.w);
        af[4]=f2bf(a1.x); af[5]=f2bf(a1.y); af[6]=f2bf(a1.z); af[7]=f2bf(a1.w);
        af[8]=f2bf(a2.x); af[9]=f2bf(a2.y); af[10]=f2bf(a2.z); af[11]=f2bf(a2.w);
        af[12]=f2bf(a3.x); af[13]=f2bf(a3.y); af[14]=f2bf(a3.z); af[15]=f2bf(a3.w);

        const float4* xp = (const float4*)(xrow + kload);
        a0 = xp[0]; a1 = xp[1]; a2 = xp[2]; a3 = xp[3];
        #pragma unroll
        for (int t = 0; t < 8; ++t)
            BN[t] = *(const v16bf*)(wrow + (size_t)(t*16) * HID + kload);

        #pragma unroll
        for (int t = 0; t < 8; ++t)
            acc[t] = __builtin_amdgcn_wmma_f32_16x16x32_bf16(
                false, af, false, BC[t], (short)0, acc[t], false, false);
    };

    for (int kc = 0; kc < HID; kc += 64) {
        // prefetch x one stage-pair ahead into near caches (WGP scope)
        if (kc + 64 < HID) {
            __builtin_prefetch(xrow + kc + 64, 0, 3);
            __builtin_prefetch(xrow + kc + 80, 0, 3);
        }
        stage(kc + 32, bbuf0, bbuf1);                       // compute kc, load kc+32
        int kn = (kc + 64 < HID) ? kc + 64 : kc + 32;       // clamp (redundant last load)
        stage(kn,      bbuf1, bbuf0);                       // compute kc+32, load kc+64
    }

    // bias
    #pragma unroll
    for (int t = 0; t < 8; ++t) {
        float bv_ = bias[t*16 + lnm];
        #pragma unroll
        for (int r = 0; r < 8; ++r) acc[t][r] += bv_;
    }

    // RoPE for q,k. D-layout: N(=head dim d) = lane%16 + 16t, M(=seq) = half*8+r.
    if (proj < 2) {
        #pragma unroll
        for (int t = 0; t < 8; ++t) {
            int d  = t*16 + lnm;
            int fi = d >> 1;
            bool even = (d & 1) == 0;
            #pragma unroll
            for (int r = 0; r < 8; ++r) {
                int sp = s_base + half*8 + r;
                float c = rope_cos[(size_t)sp * (HD/2) + fi];
                float s = rope_sin[(size_t)sp * (HD/2) + fi];
                float v = acc[t][r];
                float p = __shfl_xor(v, 1, 32);
                acc[t][r] = even ? (v * c - p * s) : (p * s + v * c);
            }
        }
    }

    const size_t BSH = (size_t)BATCH * SLEN * HD;
    const float inv_sqrt_hd = 0.08838834764831845f;   // 1/sqrt(128)

    if (proj == 0) {
        #pragma unroll
        for (int t = 0; t < 8; ++t) {
            int d = t*16 + lnm;
            #pragma unroll
            for (int r = 0; r < 8; ++r) {
                int m = m0 + half*8 + r;
                qb[(size_t)m * HD + d] = f2bf(acc[t][r] * inv_sqrt_hd);
            }
        }
    } else if (proj == 1) {
        float* outk = out + BSH;
        #pragma unroll
        for (int t = 0; t < 8; ++t) {
            int d = t*16 + lnm;
            #pragma unroll
            for (int r = 0; r < 8; ++r) {
                int m = m0 + half*8 + r;
                outk[(size_t)m * HD + d] = acc[t][r];
                kb[(size_t)m * HD + d]   = f2bf(acc[t][r]);
            }
        }
    } else {
        float* outv = out + 2 * BSH;
        #pragma unroll
        for (int t = 0; t < 8; ++t) {
            int d = t*16 + lnm;
            v8bf pk;
            #pragma unroll
            for (int r = 0; r < 8; ++r) {
                int m = m0 + half*8 + r;
                outv[(size_t)m * HD + d] = acc[t][r];
                pk[r] = f2bf(acc[t][r]);
            }
            int s0 = s_base + half*8;
            *(v8bf*)(vtb + ((size_t)bidx * HD + d) * SLEN + s0) = pk;
        }
    }
}

// ---------------------------------------------------------------------------
// Kernel 2: causal flash attention, one wave per 16 query rows.
// grid = (64, 4 batches), block = 128 (4 waves).
// q-tiles are assigned in reverse so the longest-running (largest causal
// range) tiles launch first, shrinking the tail under finite wave slots.
// ---------------------------------------------------------------------------
__global__ __launch_bounds__(128) void attn_k(
    const bf16_t* __restrict__ qb, const bf16_t* __restrict__ kb,
    const bf16_t* __restrict__ vtb, float* __restrict__ out)
{
    __shared__ __attribute__((aligned(32))) bf16_t pbuf[4][16*32];

    const int lane = threadIdx.x & 31;
    const int wid  = threadIdx.x >> 5;
    const int lnm  = lane & 15;
    const int half = lane >> 4;
    const int b    = blockIdx.y;
    const int qtile = (SLEN/16 - 1) - (blockIdx.x * 4 + wid);   // reversed
    const int q0   = qtile * 16;

    // Q A-fragments for 4 K-chunks of 32 (q already scaled by 1/sqrt(HD))
    v16bf qf[4];
    const bf16_t* qrow = qb + ((size_t)b * SLEN + q0 + lnm) * HD;
    #pragma unroll
    for (int c = 0; c < 4; ++c) {
        v8bf lo = *(const v8bf*)(qrow + c*32 + half*16);
        v8bf hi = *(const v8bf*)(qrow + c*32 + half*16 + 8);
        #pragma unroll
        for (int i = 0; i < 8; ++i) { qf[c][i] = lo[i]; qf[c][8+i] = hi[i]; }
    }

    v8f zero = {0.f,0.f,0.f,0.f,0.f,0.f,0.f,0.f};
    v8f o[8];
    #pragma unroll
    for (int t = 0; t < 8; ++t) o[t] = zero;
    float mrow[8], lrow[8];
    #pragma unroll
    for (int r = 0; r < 8; ++r) { mrow[r] = -3.0e38f; lrow[r] = 0.f; }

    const bf16_t* kbase = kb + (size_t)b * SLEN * HD;
    const bf16_t* vbase = vtb + (size_t)b * HD * SLEN;
    bf16_t* pb = pbuf[wid];

    const int kv_end = q0 + 16;
    for (int kv0 = 0; kv0 < kv_end; kv0 += 32) {
        // prefetch next iteration's K/V tiles into near caches (WGP scope)
        if (kv0 + 32 < kv_end) {
            __builtin_prefetch(kbase + (size_t)(kv0 + 32 + lane) * HD, 0, 3);
            __builtin_prefetch(kbase + (size_t)(kv0 + 32 + lane) * HD + 64, 0, 3);
            __builtin_prefetch(vbase + (size_t)lane         * SLEN + kv0 + 32, 0, 3);
            __builtin_prefetch(vbase + (size_t)(lane + 32)  * SLEN + kv0 + 32, 0, 3);
            __builtin_prefetch(vbase + (size_t)(lane + 64)  * SLEN + kv0 + 32, 0, 3);
            __builtin_prefetch(vbase + (size_t)(lane + 96)  * SLEN + kv0 + 32, 0, 3);
        }

        // S = Q * K^T : two 16-wide col tiles
        v8f s0 = zero, s1 = zero;
        #pragma unroll
        for (int c = 0; c < 4; ++c) {
            const bf16_t* kr0 = kbase + (size_t)(kv0 + lnm) * HD + c*32 + half*16;
            v16bf bf0 = *(const v16bf*)kr0;
            s0 = __builtin_amdgcn_wmma_f32_16x16x32_bf16(false, qf[c], false, bf0,
                                                         (short)0, s0, false, false);
            v16bf bf1 = *(const v16bf*)(kr0 + 16 * HD);
            s1 = __builtin_amdgcn_wmma_f32_16x16x32_bf16(false, qf[c], false, bf1,
                                                         (short)0, s1, false, false);
        }

        // V fragments for this tile are score-independent: issue loads now so the
        // softmax VALU section below hides their latency.
        v16bf varr[8];
        #pragma unroll
        for (int t = 0; t < 8; ++t) {
            int d = t*16 + lnm;
            varr[t] = *(const v16bf*)(vbase + (size_t)d * SLEN + kv0 + half*16);
        }

        // causal mask: row M = half*8+r, col N = lnm (+16)
        #pragma unroll
        for (int r = 0; r < 8; ++r) {
            int qg = q0 + half*8 + r;
            if (kv0 + lnm      > qg) s0[r] = -3.0e38f;
            if (kv0 + lnm + 16 > qg) s1[r] = -3.0e38f;
        }

        // online softmax (per-row stats live in the 16-lane half owning that row)
        float corr[8];
        #pragma unroll
        for (int r = 0; r < 8; ++r) {
            float t = fmaxf(s0[r], s1[r]);
            t = fmaxf(t, __shfl_xor(t, 1, 32));
            t = fmaxf(t, __shfl_xor(t, 2, 32));
            t = fmaxf(t, __shfl_xor(t, 4, 32));
            t = fmaxf(t, __shfl_xor(t, 8, 32));
            float mn = fmaxf(mrow[r], t);
            corr[r]  = __expf(mrow[r] - mn);
            float p0 = __expf(s0[r] - mn);
            float p1 = __expf(s1[r] - mn);
            s0[r] = p0; s1[r] = p1;
            float ps = p0 + p1;
            ps += __shfl_xor(ps, 1, 32);
            ps += __shfl_xor(ps, 2, 32);
            ps += __shfl_xor(ps, 4, 32);
            ps += __shfl_xor(ps, 8, 32);
            lrow[r] = lrow[r] * corr[r] + ps;
            mrow[r] = mn;
        }
        #pragma unroll
        for (int t = 0; t < 8; ++t)
            #pragma unroll
            for (int r = 0; r < 8; ++r) o[t][r] *= corr[r];

        // relayout P (D-frag) -> A-frag via LDS (single wave, DS ops in-order)
        #pragma unroll
        for (int r = 0; r < 8; ++r) {
            int row = half*8 + r;
            pb[row*32 + lnm]      = f2bf(s0[r]);
            pb[row*32 + 16 + lnm] = f2bf(s1[r]);
        }
        v16bf pf;
        {
            v8bf lo = *(const v8bf*)(pb + lnm*32 + half*16);
            v8bf hi = *(const v8bf*)(pb + lnm*32 + half*16 + 8);
            #pragma unroll
            for (int i = 0; i < 8; ++i) { pf[i] = lo[i]; pf[8+i] = hi[i]; }
        }

        // O += P * V
        #pragma unroll
        for (int t = 0; t < 8; ++t)
            o[t] = __builtin_amdgcn_wmma_f32_16x16x32_bf16(false, pf, false, varr[t],
                                                           (short)0, o[t], false, false);
    }

    // epilogue: normalize and store attn_result (offset 0 of d_out)
    #pragma unroll
    for (int t = 0; t < 8; ++t) {
        int d = t*16 + lnm;
        #pragma unroll
        for (int r = 0; r < 8; ++r) {
            int m = b * SLEN + q0 + half*8 + r;
            out[(size_t)m * HD + d] = o[t][r] / lrow[r];
        }
    }
}

extern "C" void kernel_launch(void* const* d_in, const int* in_sizes, int n_in,
                              void* d_out, int out_size, void* d_ws, size_t ws_size,
                              hipStream_t stream) {
    (void)in_sizes; (void)n_in; (void)out_size; (void)ws_size;
    const float* x        = (const float*)d_in[0];
    const float* wq       = (const float*)d_in[1];
    const float* bq       = (const float*)d_in[2];
    const float* wk       = (const float*)d_in[3];
    const float* bk       = (const float*)d_in[4];
    const float* wv       = (const float*)d_in[5];
    const float* bv       = (const float*)d_in[6];
    const float* rope_cos = (const float*)d_in[7];
    const float* rope_sin = (const float*)d_in[8];
    // d_in[9] attn_masked_bias: causal mask applied analytically in-kernel
    float* out = (float*)d_out;

    const size_t BSH = (size_t)BATCH * SLEN * HD;        // 2,097,152 elements
    bf16_t* qb  = (bf16_t*)d_ws;                         // 4 MB
    bf16_t* kb  = qb + BSH;                              // 4 MB
    bf16_t* vtb = kb + BSH;                              // 4 MB
    bf16_t* wb  = vtb + BSH;                             // 3 MB (3 x HD x HID bf16)

    dim3 g0((HD * HID / 4) / 256, 3);
    wcvt_k<<<g0, 256, 0, stream>>>(wq, wk, wv, wb);

    dim3 g1(3 * (BATCH * SLEN) / 16);
    qkv_rope_k<<<g1, 32, 0, stream>>>(x, wb, bq, bk, bv,
                                      rope_cos, rope_sin, out, qb, kb, vtb);

    dim3 g2(SLEN / 16 / 4, BATCH);
    attn_k<<<g2, 128, 0, stream>>>(qb, kb, vtb, out);
}
